// CrossMambaBlock_25099788878249
// MI455X (gfx1250) — compile-verified, matrix operand order checked by reading, not run on hardware
//
#include <hip/hip_runtime.h>
#include <hip/hip_bf16.h>
#include <cstdint>
#include <cstddef>

// ---------------- problem constants (match reference) ----------------
#define BATCH   4
#define SEQ     4096
#define DIMC    256
#define NST     8
#define DTR     16
#define HID     128
#define MROWS   (BATCH * SEQ)      // 16384
#define NCHUNK  64                 // chunks for parallel linear scan
#define CLEN    (SEQ / NCHUNK)     // 64

typedef __attribute__((ext_vector_type(16))) __bf16 v16bf;
typedef __attribute__((ext_vector_type(8)))  float  v8f;

// ---------------- small helpers ----------------
__device__ __forceinline__ uint16_t f2bf(float f) {
    unsigned u = __float_as_uint(f);
    unsigned r = u + 0x7fffu + ((u >> 16) & 1u);   // round-to-nearest-even
    return (uint16_t)(r >> 16);
}
__device__ __forceinline__ float silu(float x) {
    return x * (1.0f / (1.0f + __expf(-x)));
}

// ---------------- elementwise fp32 -> bf16 convert ----------------
__global__ void cvt_bf16_kernel(const float* __restrict__ in,
                                uint16_t* __restrict__ out, int n) {
    int i = blockIdx.x * blockDim.x + threadIdx.x;
    if (i < n) out[i] = f2bf(in[i]);
}

// ---------------- rmsnorm (one 256-wide row per block) -> bf16 ----------------
__global__ void rmsnorm_bf16_kernel(const float* __restrict__ x,
                                    const float* __restrict__ w,
                                    uint16_t* __restrict__ out) {
    __shared__ float red[DIMC];
    int m = blockIdx.x, d = threadIdx.x;
    float v = x[(size_t)m * DIMC + d];
    red[d] = v * v;
    __syncthreads();
    for (int s = DIMC / 2; s > 0; s >>= 1) {
        if (d < s) red[d] += red[d + s];
        __syncthreads();
    }
    float scale = 1.0f / (sqrtf(red[0]) / 16.0f + 1e-6f);   // sqrt(256)=16
    out[(size_t)m * DIMC + d] = f2bf(v * scale * w[d]);
}

// ---------------- WMMA bf16 GEMM with async-LDS weight staging --------------
// out(M,N) = A(M,K) @ W(N,K)^T [+ residual] [, bf16 mirror]
// Block = 8 waves; each wave computes a 32 x (NSUB*16) tile (MSUB=2 x NSUB
// accumulators) -> block tile = 256 rows x BN cols. The W strip (BN x K bf16,
// <=32KB) is staged into LDS once per block with GLOBAL_LOAD_ASYNC_TO_LDS_B128
// (ASYNCcnt). Per K-step: MSUB A fragments (global b128) + NSUB B fragments
// (LDS b128) are loaded into DISTINCT registers first, then MSUB*NSUB WMMAs
// issue back-to-back (single dscnt/loadcnt wait instead of one per WMMA).
// Per-lane fragment = two contiguous 16B chunks:
//   chunk0: k in [k0 + half*8, +8) ; chunk1: k in [k0+16+half*8, +8)
// (CDNA5 ISA 7.12.2 16-bit A/B layouts; C/D f32: VGPR r -> m=r+8*half, n=lane&15)
template <int NSUB>
__global__ void gemm_bf16_wmma_lds(const uint16_t* __restrict__ A,
                                   const uint16_t* __restrict__ W,
                                   float* __restrict__ out,
                                   const float* __restrict__ residual,
                                   uint16_t* __restrict__ out_bf,
                                   int M, int N, int K) {
    constexpr int BN = NSUB * 16;
    constexpr int MSUB = 2;                               // 32 rows per wave
    __shared__ __align__(16) unsigned char wtile[64 * 512];  // up to 64 rows x 512B

    const int nBlk = N / BN;
    const int tnb = blockIdx.x % nBlk;
    const int tmb = blockIdx.x / nBlk;
    const int wid = threadIdx.x >> 5;
    const int lane = threadIdx.x & 31;

    // ---- stage W strip [tnb*BN, +BN) x K into LDS via async copies ----
    const int stripBytes = BN * K * 2;                  // multiple of 4096
    const unsigned char* wsrc = (const unsigned char*)(W + (size_t)tnb * BN * K);
    unsigned ldsBase = (unsigned)(uintptr_t)wtile;
    for (int ofs = threadIdx.x * 16; ofs < stripBytes; ofs += 256 * 16) {
        unsigned long long ga = (unsigned long long)(uintptr_t)(wsrc + ofs);
        unsigned la = ldsBase + (unsigned)ofs;
        asm volatile("global_load_async_to_lds_b128 %0, %1, off"
                     :: "v"(la), "v"(ga) : "memory");
    }
    asm volatile("s_wait_asynccnt 0x0" ::: "memory");
    __syncthreads();

    const int half = lane >> 4;
    const int l15 = lane & 15;
    const int rowBase = tmb * 256 + wid * 32;            // 32 rows per wave
    const int kstride = K * 2;                           // LDS row pitch (bytes)

    union frag { v16bf v; uint4 q[2]; };
    v8f acc[MSUB][NSUB];
#pragma unroll
    for (int mi = 0; mi < MSUB; ++mi)
#pragma unroll
        for (int s = 0; s < NSUB; ++s) acc[mi][s] = (v8f){};

    for (int k0 = 0; k0 < K; k0 += 32) {
        // all fragments into distinct registers, then one wait, then WMMAs
        frag a[MSUB];
#pragma unroll
        for (int mi = 0; mi < MSUB; ++mi) {
            const uint16_t* ap =
                A + (size_t)(rowBase + mi * 16 + l15) * K + k0 + half * 8;
            a[mi].q[0] = *(const uint4*)(ap);
            a[mi].q[1] = *(const uint4*)(ap + 16);
        }
        frag b[NSUB];
#pragma unroll
        for (int s = 0; s < NSUB; ++s) {
            const unsigned char* bp =
                wtile + (size_t)(s * 16 + l15) * kstride + (k0 + half * 8) * 2;
            b[s].q[0] = *(const uint4*)(bp);
            b[s].q[1] = *(const uint4*)(bp + 32);
        }
#pragma unroll
        for (int s = 0; s < NSUB; ++s)
#pragma unroll
            for (int mi = 0; mi < MSUB; ++mi)
                acc[mi][s] = __builtin_amdgcn_wmma_f32_16x16x32_bf16(
                    /*neg_a=*/false, a[mi].v, /*neg_b=*/false, b[s].v,
                    /*c_mod=*/(short)0, acc[mi][s],
                    /*reuse_a=*/false, /*reuse_b=*/false);
    }
#pragma unroll
    for (int mi = 0; mi < MSUB; ++mi) {
#pragma unroll
        for (int s = 0; s < NSUB; ++s) {
#pragma unroll
            for (int r = 0; r < 8; ++r) {
                int m = rowBase + mi * 16 + r + 8 * half;
                int n = tnb * BN + s * 16 + l15;
                float v = acc[mi][s][r];
                if (residual) v += residual[(size_t)m * N + n];
                out[(size_t)m * N + n] = v;
                if (out_bf) out_bf[(size_t)m * N + n] = f2bf(v);
            }
        }
    }
}

// ---------------- depthwise causal / anti-causal conv + silu ----------------
// xin lives in xz with row stride 512 (columns 0..255). Output (B,L,D) fp32.
__global__ void conv_mamba_kernel(const float* __restrict__ xz,
                                  const float* __restrict__ w,
                                  const float* __restrict__ bias,
                                  float* __restrict__ xc, int backward) {
    int idx = blockIdx.x * blockDim.x + threadIdx.x;   // = ((b*SEQ)+t)*DIMC + d
    int d = idx & (DIMC - 1);
    int t = (idx / DIMC) & (SEQ - 1);
    int b = idx / (DIMC * SEQ);
    float acc = bias[d];
#pragma unroll
    for (int j = 0; j < 4; ++j) {
        int tt = backward ? (t + 3 - j) : (t - 3 + j);
        if (tt >= 0 && tt < SEQ)
            acc += w[d * 4 + j] * xz[(((size_t)b * SEQ + tt) * 512) + d];
    }
    xc[idx] = silu(acc);
}

// ---------------- delta = softplus(dt @ dtproj^T + bias) ----------------
// x_dbl: (M,32); cols 0..15 = dt. One row per block.
__global__ void delta_kernel(const float* __restrict__ xdbl,
                             const float* __restrict__ dtw,   // (256,16)
                             const float* __restrict__ dtb,
                             float* __restrict__ delta) {
    __shared__ float dt[DTR];
    int m = blockIdx.x, d = threadIdx.x;
    if (d < DTR) dt[d] = xdbl[(size_t)m * 32 + d];
    __syncthreads();
    float s = dtb[d];
#pragma unroll
    for (int k = 0; k < DTR; ++k) s += dt[k] * dtw[d * DTR + k];
    delta[(size_t)m * DIMC + d] = (s > 20.0f) ? s : log1pf(__expf(s));
}

// ---------------- chunked parallel linear scan (3 passes) ----------------
// h_t = exp(dl*A)*h_{t-1} + (dl*x)*B_t ;  y_t = <h_t, C_t> + x_t*D
__global__ void scan_pass1(const float* __restrict__ xc,
                           const float* __restrict__ delta,
                           const float* __restrict__ xdbl,  // cols 16..23 = B
                           const float* __restrict__ A_log,
                           float* __restrict__ Pout, float* __restrict__ Hout,
                           int backward) {
    int b = blockIdx.x / NCHUNK;
    int c = blockIdx.x % NCHUNK;
    int d = threadIdx.x;
    float A[NST], h[NST], P[NST];
#pragma unroll
    for (int n = 0; n < NST; ++n) {
        A[n] = -__expf(A_log[d * NST + n]);
        h[n] = 0.0f; P[n] = 1.0f;
    }
    for (int i = 0; i < CLEN; ++i) {
        int tl = c * CLEN + i;
        int t = backward ? (SEQ - 1 - tl) : tl;
        size_t m = (size_t)b * SEQ + t;
        float dl = delta[m * DIMC + d];
        float xv = xc[m * DIMC + d];
        float dx = dl * xv;
#pragma unroll
        for (int n = 0; n < NST; ++n) {
            float da = __expf(dl * A[n]);
            float Bn = xdbl[m * 32 + 16 + n];
            h[n] = da * h[n] + dx * Bn;
            P[n] *= da;
        }
    }
    size_t base = (((size_t)b * NCHUNK + c) * DIMC + d) * NST;
#pragma unroll
    for (int n = 0; n < NST; ++n) { Pout[base + n] = P[n]; Hout[base + n] = h[n]; }
}

__global__ void scan_pass2(const float* __restrict__ P,
                           const float* __restrict__ Hloc,
                           float* __restrict__ Hin) {
    int b = blockIdx.x, d = threadIdx.x;
    float h[NST];
#pragma unroll
    for (int n = 0; n < NST; ++n) h[n] = 0.0f;
    for (int c = 0; c < NCHUNK; ++c) {
        size_t base = (((size_t)b * NCHUNK + c) * DIMC + d) * NST;
#pragma unroll
        for (int n = 0; n < NST; ++n) {
            Hin[base + n] = h[n];
            h[n] = P[base + n] * h[n] + Hloc[base + n];
        }
    }
}

__global__ void scan_pass3(const float* __restrict__ xc,
                           const float* __restrict__ delta,
                           const float* __restrict__ xdbl,  // 16..23=B, 24..31=C
                           const float* __restrict__ A_log,
                           const float* __restrict__ Hin,
                           const float* __restrict__ Dp,
                           float* __restrict__ y, int backward) {
    int b = blockIdx.x / NCHUNK;
    int c = blockIdx.x % NCHUNK;
    int d = threadIdx.x;
    float A[NST], h[NST];
    size_t base = (((size_t)b * NCHUNK + c) * DIMC + d) * NST;
#pragma unroll
    for (int n = 0; n < NST; ++n) {
        A[n] = -__expf(A_log[d * NST + n]);
        h[n] = Hin[base + n];
    }
    float Dd = Dp[d];
    for (int i = 0; i < CLEN; ++i) {
        int tl = c * CLEN + i;
        int t = backward ? (SEQ - 1 - tl) : tl;
        size_t m = (size_t)b * SEQ + t;
        float dl = delta[m * DIMC + d];
        float xv = xc[m * DIMC + d];
        float dx = dl * xv;
        float acc = 0.0f;
#pragma unroll
        for (int n = 0; n < NST; ++n) {
            float da = __expf(dl * A[n]);
            float Bn = xdbl[m * 32 + 16 + n];
            float Cn = xdbl[m * 32 + 24 + n];
            h[n] = da * h[n] + dx * Bn;
            acc += h[n] * Cn;
        }
        y[m * DIMC + d] = acc + xv * Dd;
    }
}

// ---------------- y = rmsnorm(0.5*(yf+yb)) * w * silu(z) -> bf16 ----------------
__global__ void combine_norm_gate(const float* __restrict__ yf,
                                  const float* __restrict__ yb,
                                  const float* __restrict__ wn,
                                  const float* __restrict__ xz,  // z at col 256+d
                                  uint16_t* __restrict__ out) {
    __shared__ float red[DIMC];
    int m = blockIdx.x, d = threadIdx.x;
    float y = 0.5f * (yf[(size_t)m * DIMC + d] + yb[(size_t)m * DIMC + d]);
    red[d] = y * y;
    __syncthreads();
    for (int s = DIMC / 2; s > 0; s >>= 1) {
        if (d < s) red[d] += red[d + s];
        __syncthreads();
    }
    float scale = 1.0f / (sqrtf(red[0]) / 16.0f + 1e-6f);
    float z = xz[(size_t)m * 512 + 256 + d];
    out[(size_t)m * DIMC + d] = f2bf(y * scale * wn[d] * silu(z));
}

// ---------------- FFN depthwise conv (k=3, pad 1,1) + silu -> bf16 ------------
__global__ void dwconv_ffn_kernel(const float* __restrict__ m1,
                                  const float* __restrict__ w,   // (128,1,3)
                                  const float* __restrict__ bias,
                                  uint16_t* __restrict__ out) {
    int idx = blockIdx.x * blockDim.x + threadIdx.x;   // = ((b*SEQ)+t)*HID + c
    int c = idx & (HID - 1);
    int t = (idx / HID) & (SEQ - 1);
    int b = idx / (HID * SEQ);
    float acc = bias[c];
#pragma unroll
    for (int j = 0; j < 3; ++j) {
        int tt = t - 1 + j;
        if (tt >= 0 && tt < SEQ)
            acc += w[c * 3 + j] * m1[(((size_t)b * SEQ + tt) * HID) + c];
    }
    out[idx] = f2bf(silu(acc));
}

// ---------------- host side ----------------
static inline char* carve(char*& p, size_t bytes) {
    char* r = p;
    p += (bytes + 255) & ~(size_t)255;
    return r;
}

extern "C" void kernel_launch(void* const* d_in, const int* in_sizes, int n_in,
                              void* d_out, int out_size, void* d_ws, size_t ws_size,
                              hipStream_t stream) {
    // inputs per setup_inputs() order
    const float* x0        = (const float*)d_in[0];
    const float* x1        = (const float*)d_in[1];
    const float* w_norm0   = (const float*)d_in[2];
    const float* w_norm1   = (const float*)d_in[3];
    const float* in_proj_w = (const float*)d_in[4];
    const float* conv_w_f  = (const float*)d_in[5];
    const float* conv_b_f  = (const float*)d_in[6];
    const float* xproj_w_f = (const float*)d_in[7];
    const float* dtproj_w_f= (const float*)d_in[8];
    const float* dtproj_b_f= (const float*)d_in[9];
    const float* A_log_f   = (const float*)d_in[10];
    const float* D_f       = (const float*)d_in[11];
    const float* conv_w_bw = (const float*)d_in[12];
    const float* conv_b_bw = (const float*)d_in[13];
    const float* xproj_w_bw= (const float*)d_in[14];
    const float* dtproj_w_bw=(const float*)d_in[15];
    const float* dtproj_b_bw=(const float*)d_in[16];
    const float* A_log_bw  = (const float*)d_in[17];
    const float* D_bw      = (const float*)d_in[18];
    const float* norm_y_w  = (const float*)d_in[19];
    const float* out_proj_w= (const float*)d_in[20];
    const float* fc1_w     = (const float*)d_in[21];
    const float* dw_w      = (const float*)d_in[22];
    const float* dw_b      = (const float*)d_in[23];
    const float* fc2_w     = (const float*)d_in[24];
    float* outp = (float*)d_out;

    // workspace carve
    char* p = (char*)d_ws;
    uint16_t* h0b   = (uint16_t*)carve(p, (size_t)MROWS * DIMC * 2);
    uint16_t* h1b   = (uint16_t*)carve(p, (size_t)MROWS * DIMC * 2);
    uint16_t* wb_in = (uint16_t*)carve(p, (size_t)512 * 256 * 2);
    uint16_t* wb_xf = (uint16_t*)carve(p, (size_t)32 * 256 * 2);
    uint16_t* wb_xb = (uint16_t*)carve(p, (size_t)32 * 256 * 2);
    uint16_t* wb_op = (uint16_t*)carve(p, (size_t)256 * 256 * 2);
    uint16_t* wb_f1 = (uint16_t*)carve(p, (size_t)128 * 256 * 2);
    uint16_t* wb_f2 = (uint16_t*)carve(p, (size_t)256 * 128 * 2);
    float* xz     = (float*)carve(p, (size_t)MROWS * 512 * 4);
    float* xdbl_f = (float*)carve(p, (size_t)MROWS * 32 * 4);
    float* xdbl_b = (float*)carve(p, (size_t)MROWS * 32 * 4);
    float* delta_f= (float*)carve(p, (size_t)MROWS * DIMC * 4);
    float* delta_b= (float*)carve(p, (size_t)MROWS * DIMC * 4);
    float* xc_f   = (float*)carve(p, (size_t)MROWS * DIMC * 4);
    float* xc_b   = (float*)carve(p, (size_t)MROWS * DIMC * 4);
    float* y_f    = (float*)carve(p, (size_t)MROWS * DIMC * 4);
    float* y_b    = (float*)carve(p, (size_t)MROWS * DIMC * 4);
    size_t chunkEl = (size_t)BATCH * NCHUNK * DIMC * NST;
    float* Pf   = (float*)carve(p, chunkEl * 4);
    float* Hlf  = (float*)carve(p, chunkEl * 4);
    float* Hinf = (float*)carve(p, chunkEl * 4);
    float* Pb   = (float*)carve(p, chunkEl * 4);
    float* Hlb  = (float*)carve(p, chunkEl * 4);
    float* Hinb = (float*)carve(p, chunkEl * 4);
    uint16_t* ybf16 = (uint16_t*)carve(p, (size_t)MROWS * DIMC * 2);
    float*    xbuf  = (float*)carve(p, (size_t)MROWS * DIMC * 4);
    uint16_t* xb16  = (uint16_t*)carve(p, (size_t)MROWS * DIMC * 2);
    float*    m1    = (float*)carve(p, (size_t)MROWS * HID * 4);
    uint16_t* m1c   = (uint16_t*)carve(p, (size_t)MROWS * HID * 2);
    (void)ws_size; (void)n_in; (void)in_sizes; (void)out_size;

    const int T = 256;
    auto cvt = [&](const float* src, uint16_t* dst, int n) {
        cvt_bf16_kernel<<<(n + T - 1) / T, T, 0, stream>>>(src, dst, n);
    };
    auto gemm = [&](const uint16_t* A, const uint16_t* W, float* out,
                    const float* res, uint16_t* obf, int N, int K) {
        if (N % 64 == 0)
            gemm_bf16_wmma_lds<4><<<(MROWS / 256) * (N / 64), 256, 0, stream>>>(
                A, W, out, res, obf, MROWS, N, K);
        else
            gemm_bf16_wmma_lds<2><<<(MROWS / 256) * (N / 32), 256, 0, stream>>>(
                A, W, out, res, obf, MROWS, N, K);
    };

    // 1) rmsnorm both streams -> bf16
    rmsnorm_bf16_kernel<<<MROWS, T, 0, stream>>>(x0, w_norm0, h0b);
    rmsnorm_bf16_kernel<<<MROWS, T, 0, stream>>>(x1, w_norm1, h1b);

    // 2) weight conversions (tiny, once per call)
    cvt(in_proj_w, wb_in, 512 * 256);
    cvt(xproj_w_f, wb_xf, 32 * 256);
    cvt(xproj_w_bw, wb_xb, 32 * 256);
    cvt(out_proj_w, wb_op, 256 * 256);
    cvt(fc1_w, wb_f1, 128 * 256);
    cvt(fc2_w, wb_f2, 256 * 128);

    // 3) xz = h0 @ in_proj^T  (xin = cols 0..255, z = cols 256..511)
    gemm(h0b, wb_in, xz, nullptr, nullptr, 512, 256);

    // 4) x_dbl (dt|B|C) for both directions (flip-free: rows stay in orig order)
    gemm(h1b, wb_xf, xdbl_f, nullptr, nullptr, 32, 256);
    gemm(h1b, wb_xb, xdbl_b, nullptr, nullptr, 32, 256);

    // 5) depthwise conv + silu (causal fwd / anti-causal bwd, unflipped layout)
    int convBlocks = (MROWS * DIMC) / T;
    conv_mamba_kernel<<<convBlocks, T, 0, stream>>>(xz, conv_w_f, conv_b_f, xc_f, 0);
    conv_mamba_kernel<<<convBlocks, T, 0, stream>>>(xz, conv_w_bw, conv_b_bw, xc_b, 1);

    // 6) delta = softplus(dt @ dtproj^T + bias)
    delta_kernel<<<MROWS, T, 0, stream>>>(xdbl_f, dtproj_w_f, dtproj_b_f, delta_f);
    delta_kernel<<<MROWS, T, 0, stream>>>(xdbl_b, dtproj_w_bw, dtproj_b_bw, delta_b);

    // 7) chunked parallel selective scan, forward & backward
    scan_pass1<<<BATCH * NCHUNK, T, 0, stream>>>(xc_f, delta_f, xdbl_f, A_log_f, Pf, Hlf, 0);
    scan_pass2<<<BATCH, T, 0, stream>>>(Pf, Hlf, Hinf);
    scan_pass3<<<BATCH * NCHUNK, T, 0, stream>>>(xc_f, delta_f, xdbl_f, A_log_f, Hinf, D_f, y_f, 0);
    scan_pass1<<<BATCH * NCHUNK, T, 0, stream>>>(xc_b, delta_b, xdbl_b, A_log_bw, Pb, Hlb, 1);
    scan_pass2<<<BATCH, T, 0, stream>>>(Pb, Hlb, Hinb);
    scan_pass3<<<BATCH * NCHUNK, T, 0, stream>>>(xc_b, delta_b, xdbl_b, A_log_bw, Hinb, D_bw, y_b, 1);

    // 8) y = rmsnorm(0.5*(yf+yb)) * w * silu(z) -> bf16
    combine_norm_gate<<<MROWS, T, 0, stream>>>(y_f, y_b, norm_y_w, xz, ybf16);

    // 9) x = y @ out_proj^T + x0 (residual + bf16 mirror fused in epilogue)
    gemm(ybf16, wb_op, xbuf, x0, xb16, 256, 256);

    // 10) FFN: m1 = x @ fc1^T ; silu(dwconv(m1)) ; out = m2 @ fc2^T + x
    gemm(xb16, wb_f1, m1, nullptr, nullptr, 128, 256);
    dwconv_ffn_kernel<<<(MROWS * HID) / T, T, 0, stream>>>(m1, dw_w, dw_b, m1c);
    gemm(m1c, wb_f2, outp, xbuf, nullptr, 256, 128);
}
